// LibRNN_6279242186941
// MI455X (gfx1250) — matrix operand order repaired and checked by reading, not compiled, not run on hardware
//
#include <hip/hip_runtime.h>
#include <hip/hip_bf16.h>
#include <cstddef>

// Problem constants (match reference): B=1024, T=512, H=64, L=3
#define B_ 1024
#define T_ 512
#define H_ 64
#define L_ 3

typedef __bf16        v16bf  __attribute__((ext_vector_type(16)));
typedef float         v8f    __attribute__((ext_vector_type(8)));
typedef unsigned int  u32x4  __attribute__((ext_vector_type(4)));

union BFrag {
  v16bf v;
  u32x4 q[2];
};

// Native tanh (CDNA5 TRANS op) if exposed.
#if __has_builtin(__builtin_amdgcn_tanhf)
#define TANH_(x) __builtin_amdgcn_tanhf(x)
#elif __has_builtin(__builtin_amdgcn_tanh_f32)
#define TANH_(x) __builtin_amdgcn_tanh_f32(x)
#else
#define TANH_(x) tanhf(x)
#endif

__device__ __forceinline__ v8f wmma_bf16(v16bf a, v16bf b, v8f c) {
  // D = A(16x32 bf16) * B(32x16 bf16) + C(16x16 f32)
  return __builtin_amdgcn_wmma_f32_16x16x32_bf16(false, a, false, b, (short)0, c,
                                                 false, false);
}

// Load an A-operand fragment (16x32 bf16, K-tile kt) from a 16x64 bf16 LDS tile.
// ISA A layout: lane (half=lane>>4, m=lane&15); elems 0..7 -> K = kt*32 + half*8 + j,
// elems 8..15 -> K = kt*32 + 16 + half*8 + (j-8). Both chunks are 16B contiguous.
__device__ __forceinline__ v16bf load_tileA(const __bf16* tl, int kt, int l15, int half) {
  BFrag f;
  const int k0 = kt * 32 + half * 8;
  f.q[0] = *(const u32x4*)(tl + l15 * H_ + k0);
  f.q[1] = *(const u32x4*)(tl + l15 * H_ + k0 + 16);
  return f.v;
}

// Build an A fragment from one global fp32 row of x (this lane's batch row, step t).
__device__ __forceinline__ v16bf load_xA(const float* __restrict__ xr, int kt, int half) {
  v16bf a;
  const int k0 = kt * 32 + half * 8;
#pragma unroll
  for (int j = 0; j < 8; ++j) {
    a[j]     = (__bf16)xr[k0 + j];
    a[8 + j] = (__bf16)xr[k0 + 16 + j];
  }
  return a;
}

// One RNN layer for one timestep. Each wave owns one 16-column N-tile (nt = wave).
// Reads h_prev from `told`, writes h = tanh(in*Wih^T + h_prev*Whh^T + bias) to `tnew`.
// Two independent accumulator chains (input path / recurrent path) keep the WMMA
// dependency depth at 2 instead of 4; single barrier (double-buffered tiles).
__device__ __forceinline__ void layer_step(const v16bf* in, const __bf16* told,
                                           __bf16* tnew,
                                           const v16bf* wih, const v16bf* whh,
                                           float bias, int lane, int wave) {
  const int half = lane >> 4, l15 = lane & 15;
  v16bf rec0 = load_tileA(told, 0, l15, half);
  v16bf rec1 = load_tileA(told, 1, l15, half);
  v8f a, b;
#pragma unroll
  for (int r = 0; r < 8; ++r) { a[r] = bias; b[r] = 0.0f; }
  a = wmma_bf16(in[0], wih[0], a);   // input path, K-tile 0
  b = wmma_bf16(rec0,  whh[0], b);   // recurrent path, K-tile 0 (independent chain)
  a = wmma_bf16(in[1], wih[1], a);   // input path, K-tile 1
  b = wmma_bf16(rec1,  whh[1], b);   // recurrent path, K-tile 1
  a = a + b;
  const int n0 = wave * 16 + l15;    // C layout: N = lane&15 within this wave's N-tile
#pragma unroll
  for (int r = 0; r < 8; ++r)        // C layout: M = r + 8*half
    tnew[(r + 8 * half) * H_ + n0] = (__bf16)TANH_(a[r]);
  __syncthreads();                   // new h visible before anyone consumes it
}

__global__ __launch_bounds__(128)
void rnn3_fused_wmma(const float* __restrict__ x,
                     const float* __restrict__ Wih, const float* __restrict__ Whh,
                     const float* __restrict__ bih, const float* __restrict__ bhh,
                     const float* __restrict__ fcw, const float* __restrict__ fcb,
                     float* __restrict__ out) {
  // Per-layer hidden state tiles, double-buffered by step parity:
  // 16 batch rows x 64 cols bf16 each. 12 KB total LDS.
  __shared__ __bf16 tile[L_][2][16 * H_];

  const int tid = threadIdx.x, wave = tid >> 5, lane = tid & 31;
  const int half = lane >> 4, l15 = lane & 15;
  const int wgBase = blockIdx.x * 16;               // this block's batch rows

  { // h0 = 0 (parity-1 buffers are read at t=0; zero everything, it's cheap)
    unsigned int* tp = (unsigned int*)&tile[0][0][0];
    for (int i = tid; i < L_ * 2 * 16 * H_ / 2; i += 128) tp[i] = 0u;
  }

  // Stage this wave's N-tile of every weight matrix as WMMA B fragments in registers.
  // B layout: elem j of lane -> W[K = kt*32 + half*8 + (j&7) + (j>>3)*16][N = nt*16 + l15],
  // and since out = in * W^T, B[K=i][N=n] = W[n*H + i].  96 VGPRs total.
  v16bf wf[2 * L_][2];   // [layer*2 + {0:Wih,1:Whh}][kt]
#pragma unroll
  for (int m = 0; m < 2 * L_; ++m) {
    const float* W = ((m & 1) ? Whh : Wih) + (m >> 1) * H_ * H_;
#pragma unroll
    for (int kt = 0; kt < 2; ++kt) {
      v16bf f;
#pragma unroll
      for (int j = 0; j < 16; ++j)
        f[j] = (__bf16)W[(wave * 16 + l15) * H_ + kt * 32 + half * 8 + (j & 7) + ((j >> 3) << 4)];
      wf[m][kt] = f;
    }
  }

  // Combined bias, one scalar per layer for this lane's output column.
  float bias[L_];
#pragma unroll
  for (int l = 0; l < L_; ++l) {
    const int n = wave * 16 + l15;
    bias[l] = bih[l * H_ + n] + bhh[l * H_ + n];
  }

  const float* xbase = x + ((size_t)(wgBase + l15) * T_) * H_;
  __syncthreads();

#pragma unroll 1
  for (int t = 0; t < T_; ++t) {
    const int p = t & 1, q = p ^ 1;   // write parity p, recurrent reads parity q
    const float* xr = xbase + (size_t)t * H_;
    if (t + 1 < T_) {  // hide next step's only global-memory latency
      __builtin_prefetch(xr + H_, 0, 0);
      __builtin_prefetch(xr + H_ + 32, 0, 0);
    }
    v16bf in0[2] = { load_xA(xr, 0, half), load_xA(xr, 1, half) };
    layer_step(in0, &tile[0][q][0], &tile[0][p][0], wf[0], wf[1], bias[0], lane, wave);

    v16bf in1[2] = { load_tileA(&tile[0][p][0], 0, l15, half),
                     load_tileA(&tile[0][p][0], 1, l15, half) };
    layer_step(in1, &tile[1][q][0], &tile[1][p][0], wf[2], wf[3], bias[1], lane, wave);

    v16bf in2[2] = { load_tileA(&tile[1][p][0], 0, l15, half),
                     load_tileA(&tile[1][p][0], 1, l15, half) };
    layer_step(in2, &tile[2][q][0], &tile[2][p][0], wf[4], wf[5], bias[2], lane, wave);
  }

  // FC on last timestep: out[b] = h3[b,:] . fc_w + fc_b   (16 rows per block)
  if (tid < 16) {
    const int pf = (T_ - 1) & 1;
    float s = fcb[0];
    const __bf16* h3 = &tile[2][pf][tid * H_];
#pragma unroll 1
    for (int n = 0; n < H_; ++n) s += (float)h3[n] * fcw[n];
    out[wgBase + tid] = s;
  }
}

extern "C" void kernel_launch(void* const* d_in, const int* in_sizes, int n_in,
                              void* d_out, int out_size, void* d_ws, size_t ws_size,
                              hipStream_t stream) {
  const float* x   = (const float*)d_in[0];
  const float* Wih = (const float*)d_in[1];
  const float* Whh = (const float*)d_in[2];
  const float* bih = (const float*)d_in[3];
  const float* bhh = (const float*)d_in[4];
  const float* fcw = (const float*)d_in[5];
  const float* fcb = (const float*)d_in[6];
  float* out = (float*)d_out;

  dim3 grid(B_ / 16);   // 64 workgroups, one 16-row batch tile each
  dim3 block(128);      // 4 wave32s, one 16-column N-tile per wave
  hipLaunchKernelGGL(rnn3_fused_wmma, grid, block, 0, stream,
                     x, Wih, Whh, bih, bhh, fcw, fcb, out);
}
